// discriminator_73572789780724
// MI455X (gfx1250) — matrix operand order repaired and checked by reading, not compiled, not run on hardware
//
#include <hip/hip_runtime.h>
#include <hip/hip_bf16.h>
#include <math.h>
#include <stdint.h>

// Problem constants
#define Z1    128
#define CONDD 12
#define HID   1024
#define NSTEP 32
#define BSZ   4096
#define G3    (3*HID)       // 3072
#define CPAD  32            // cond K padded to one WMMA chunk

typedef __bf16 bf16_t;
typedef __attribute__((ext_vector_type(16))) __bf16 v16bf;
typedef __attribute__((ext_vector_type(8)))  __bf16 v8bfv;
typedef __attribute__((ext_vector_type(8)))  float   v8f;

// ---- WMMA helpers -----------------------------------------------------------

__device__ __forceinline__ v8f wmma_bf16(v16bf a, v16bf b, v8f c) {
  // D = A(16x32 bf16) * B(32x16 bf16) + C(16x16 f32)
  return __builtin_amdgcn_wmma_f32_16x16x32_bf16(
      /*neg_a=*/false, a, /*neg_b=*/false, b,
      /*c_mod=*/(short)0, c, /*reuse_a=*/false, /*reuse_b=*/false);
}

// A-fragment: 16-bit A 16x32 — lane holds two runs of 8 consecutive K values,
// second run is +16 in K. p0 -> elements 0..7, p0+16 -> elements 8..15.
__device__ __forceinline__ v16bf load_a_frag(const bf16_t* p0) {
  union { v16bf v; v8bfv h[2]; } u;
  u.h[0] = *(const v8bfv*)(p0);
  u.h[1] = *(const v8bfv*)(p0 + 16);
  return u.v;
}

// B-fragment: 32x16 B — lane holds two runs of 8 consecutive K values, second run +16 in K.
__device__ __forceinline__ v16bf load_b_frag(const bf16_t* p0) {
  union { v16bf v; v8bfv h[2]; } u;
  u.h[0] = *(const v8bfv*)(p0);
  u.h[1] = *(const v8bfv*)(p0 + 16);
  return u.v;
}

__device__ __forceinline__ float sigmoidf_(float x) {
  return 1.0f / (1.0f + __expf(-x));
}

// ---- Kernel 0: convert / repack weights to bf16 -----------------------------
__global__ __launch_bounds__(256)
void k_cvt(const float* __restrict__ Wih, const float* __restrict__ Whh,
           const float* __restrict__ Winit, const float* __restrict__ Wout,
           const float* __restrict__ z,
           bf16_t* __restrict__ Whh_b, bf16_t* __restrict__ Wihz,
           bf16_t* __restrict__ Wihc,  bf16_t* __restrict__ Winit_b,
           bf16_t* __restrict__ Wout_b, bf16_t* __restrict__ zb) {
  int idx = blockIdx.x * 256 + threadIdx.x;
  if (idx < G3 * HID) Whh_b[idx] = (bf16_t)Whh[idx];
  if (idx < G3 * Z1) {
    int g = idx >> 7, k = idx & 127;
    Wihz[idx] = (bf16_t)Wih[g * (Z1 + CONDD) + CONDD + k];
  }
  if (idx < G3 * CPAD) {
    int g = idx >> 5, c = idx & 31;
    Wihc[idx] = (c < CONDD) ? (bf16_t)Wih[g * (Z1 + CONDD) + c] : (bf16_t)0.0f;
  }
  if (idx < HID * Z1) Winit_b[idx] = (bf16_t)Winit[idx];
  if (idx < 16 * HID) {
    int r = idx >> 10, k = idx & 1023;
    Wout_b[idx] = (r < CONDD) ? (bf16_t)Wout[r * HID + k] : (bf16_t)0.0f;
  }
  if (idx < BSZ * Z1) zb[idx] = (bf16_t)z[idx];
}

// ---- Kernel 1: h0 = tanh(z @ W_init^T + b_init) -----------------------------
__global__ __launch_bounds__(256)
void k_init(const bf16_t* __restrict__ zb, const bf16_t* __restrict__ Wi,
            const float* __restrict__ b_init,
            float* __restrict__ hf, bf16_t* __restrict__ hb) {
  int wave = threadIdx.x >> 5, lane = threadIdx.x & 31;
  int lane16 = lane & 15, half = lane >> 4;
  int tile = blockIdx.x * 8 + wave;          // 0 .. 16383
  int mbase = (tile >> 6) * 16;              // 256 m-tiles
  int nbase = (tile & 63) * 16;              // 64 n-tiles
  const bf16_t* Arow = zb + (size_t)(mbase + lane16) * Z1;
  const bf16_t* Brow = Wi + (size_t)(nbase + lane16) * Z1;
  v8f acc = {};
  for (int kk = 0; kk < Z1 / 32; ++kk) {
    v16bf a = load_a_frag(Arow + kk * 32 + half * 8);
    v16bf b = load_b_frag(Brow + kk * 32 + half * 8);
    acc = wmma_bf16(a, b, acc);
  }
  int n = nbase + lane16;
  float bias = b_init[n];
  for (int i = 0; i < 8; ++i) {
    int m = mbase + i + 8 * half;
    float v = tanhf(acc[i] + bias);
    hf[(size_t)m * HID + n] = v;
    hb[(size_t)m * HID + n] = (bf16_t)v;
  }
}

// ---- Kernel 2: giz = z @ W_ih[:,12:]^T + b_ih  (time-invariant part) --------
__global__ __launch_bounds__(256)
void k_giz(const bf16_t* __restrict__ zb, const bf16_t* __restrict__ Wihz,
           const float* __restrict__ b_ih, float* __restrict__ giz) {
  int wave = threadIdx.x >> 5, lane = threadIdx.x & 31;
  int lane16 = lane & 15, half = lane >> 4;
  int tile = blockIdx.x * 8 + wave;          // 0 .. 49151
  int mbase = (tile / 192) * 16;
  int nbase = (tile % 192) * 16;
  const bf16_t* Arow = zb + (size_t)(mbase + lane16) * Z1;
  const bf16_t* Brow = Wihz + (size_t)(nbase + lane16) * Z1;
  v8f acc = {};
  for (int kk = 0; kk < Z1 / 32; ++kk) {
    v16bf a = load_a_frag(Arow + kk * 32 + half * 8);
    v16bf b = load_b_frag(Brow + kk * 32 + half * 8);
    acc = wmma_bf16(a, b, acc);
  }
  int n = nbase + lane16;
  float bias = b_ih[n];
  for (int i = 0; i < 8; ++i) {
    int m = mbase + i + 8 * half;
    giz[(size_t)m * G3 + n] = acc[i] + bias;
  }
}

// ---- GRU pointwise update for one gate-column tile --------------------------
__device__ __forceinline__ void gru_pointwise(
    int bbase, int half, int j, const v8f& ar, const v8f& az,
    const v8f& ahn, const v8f& ain, const float* __restrict__ b_hh,
    const float* __restrict__ giz, float* __restrict__ hf,
    bf16_t* __restrict__ hb) {
  float bhr = b_hh[j], bhz = b_hh[HID + j], bhn = b_hh[2 * HID + j];
  for (int i = 0; i < 8; ++i) {
    int b = bbase + i + 8 * half;
    const float* g = giz + (size_t)b * G3;
    float r   = sigmoidf_(ar[i] + g[j] + bhr);
    float u   = sigmoidf_(az[i] + g[HID + j] + bhz);
    float in_ = ain[i] + g[2 * HID + j];           // b_ih folded into giz
    float hn_ = ahn[i] + bhn;
    float nn  = tanhf(in_ + r * hn_);
    size_t hi = (size_t)b * HID + j;
    float ho  = hf[hi];
    float hnew = (1.0f - u) * nn + u * ho;
    hf[hi] = hnew;
    hb[hi] = (bf16_t)hnew;
  }
}

// ---- Kernel 3: one GRU step -------------------------------------------------
// Block = 16 batch rows x full 3072 gate cols; 8 waves, wave owns a 128-wide
// hidden slice so gate cols (j, j+1024, j+2048) stay wave-local.
// Old h (bf16) staged to LDS via async-to-LDS; A frags from LDS, W_hh from L2.
// j-tiles processed in pairs: one A fragment feeds 6 WMMAs.
__global__ __launch_bounds__(256)
void k_gru(const float* __restrict__ cond, const float* __restrict__ b_hh,
           const float* __restrict__ giz, const bf16_t* __restrict__ Whh,
           const bf16_t* __restrict__ Wihc,
           float* __restrict__ hf, bf16_t* __restrict__ hb, int t) {
  __shared__ bf16_t sh[16 * HID];            // 32 KB old-h tile (bf16)
  int bbase = blockIdx.x * 16;

  { // async stage: hb[bbase:bbase+16, :] -> LDS (ASYNCcnt-tracked DMA path)
    unsigned lds_base = (unsigned)(uintptr_t)(&sh[0]);
    const bf16_t* gsrc = hb + (size_t)bbase * HID;
    for (int i = threadIdx.x; i < (16 * HID) / 8; i += 256) {
      unsigned loff = lds_base + (unsigned)i * 16u;   // per-lane LDS byte addr
      const bf16_t* gp = gsrc + (size_t)i * 8;        // 16 bytes per lane
      asm volatile("global_load_async_to_lds_b128 %0, %1, off"
                   :: "v"(loff), "v"(gp)
                   : "memory");
    }
    asm volatile("s_wait_asynccnt 0x0" ::: "memory");
  }
  __syncthreads();

  int wave = threadIdx.x >> 5, lane = threadIdx.x & 31;
  int lane16 = lane & 15, half = lane >> 4;
  int brow = bbase + lane16;

  // cond A-fragment (K padded 12 -> 32), feed_t = (t==0) ? 0 : cond[:, t-1, :]
  v16bf a_c = {};
  if (t > 0) {
    const float* cp = cond + (size_t)brow * (NSTEP * CONDD) + (t - 1) * CONDD;
    if (half == 0) {
      for (int i = 0; i < 8; ++i) a_c[i] = (bf16_t)cp[i];        // K 0..7
    } else {
      for (int i = 0; i < 4; ++i) a_c[i] = (bf16_t)cp[8 + i];    // K 8..11
    }
  }

  const bf16_t* Arow = sh + (size_t)lane16 * HID;

  for (int jp = 0; jp < 4; ++jp) {           // pairs of j-tiles
    int j0 = wave * 128 + (2 * jp) * 16 + lane16;
    int j1 = j0 + 16;
    const bf16_t* Wr0 = Whh + (size_t)j0 * HID;
    const bf16_t* Wz0 = Whh + (size_t)(HID + j0) * HID;
    const bf16_t* Wn0 = Whh + (size_t)(2 * HID + j0) * HID;
    const bf16_t* Wr1 = Whh + (size_t)j1 * HID;
    const bf16_t* Wz1 = Whh + (size_t)(HID + j1) * HID;
    const bf16_t* Wn1 = Whh + (size_t)(2 * HID + j1) * HID;

    v8f ar0 = {}, az0 = {}, ahn0 = {}, ain0 = {};
    v8f ar1 = {}, az1 = {}, ahn1 = {}, ain1 = {};
    for (int kk = 0; kk < HID / 32; ++kk) {
      v16bf a = load_a_frag(Arow + kk * 32 + half * 8);   // ds_load_b128 x2
      int kb = kk * 32 + half * 8;
      v16bf br0 = load_b_frag(Wr0 + kb);
      v16bf bz0 = load_b_frag(Wz0 + kb);
      v16bf bn0 = load_b_frag(Wn0 + kb);
      v16bf br1 = load_b_frag(Wr1 + kb);
      v16bf bz1 = load_b_frag(Wz1 + kb);
      v16bf bn1 = load_b_frag(Wn1 + kb);
      ar0  = wmma_bf16(a, br0, ar0);
      az0  = wmma_bf16(a, bz0, az0);
      ahn0 = wmma_bf16(a, bn0, ahn0);
      ar1  = wmma_bf16(a, br1, ar1);
      az1  = wmma_bf16(a, bz1, az1);
      ahn1 = wmma_bf16(a, bn1, ahn1);
    }
    { // cond K-chunk: adds i-gate feed contribution (i_n kept separate)
      int kb = half * 8;
      v16bf cr0 = load_b_frag(Wihc + (size_t)j0 * CPAD + kb);
      v16bf cz0 = load_b_frag(Wihc + (size_t)(HID + j0) * CPAD + kb);
      v16bf cn0 = load_b_frag(Wihc + (size_t)(2 * HID + j0) * CPAD + kb);
      v16bf cr1 = load_b_frag(Wihc + (size_t)j1 * CPAD + kb);
      v16bf cz1 = load_b_frag(Wihc + (size_t)(HID + j1) * CPAD + kb);
      v16bf cn1 = load_b_frag(Wihc + (size_t)(2 * HID + j1) * CPAD + kb);
      ar0  = wmma_bf16(a_c, cr0, ar0);
      az0  = wmma_bf16(a_c, cz0, az0);
      ain0 = wmma_bf16(a_c, cn0, ain0);
      ar1  = wmma_bf16(a_c, cr1, ar1);
      az1  = wmma_bf16(a_c, cz1, az1);
      ain1 = wmma_bf16(a_c, cn1, ain1);
    }
    gru_pointwise(bbase, half, j0, ar0, az0, ahn0, ain0, b_hh, giz, hf, hb);
    gru_pointwise(bbase, half, j1, ar1, az1, ahn1, ain1, b_hh, giz, hf, hb);
  }
}

// ---- Kernel 4: x_t = sigmoid(h @ W_out^T + b_out), write out[:, t, :] -------
__global__ __launch_bounds__(256)
void k_out(const bf16_t* __restrict__ hb, const bf16_t* __restrict__ Wob,
           const float* __restrict__ b_out, float* __restrict__ out, int t) {
  int wave = threadIdx.x >> 5, lane = threadIdx.x & 31;
  int lane16 = lane & 15, half = lane >> 4;
  int mbase = (blockIdx.x * 8 + wave) * 16;   // 256 m-tiles
  const bf16_t* Arow = hb + (size_t)(mbase + lane16) * HID;
  const bf16_t* Brow = Wob + (size_t)lane16 * HID;
  v8f acc = {};
  for (int kk = 0; kk < HID / 32; ++kk) {
    v16bf a = load_a_frag(Arow + kk * 32 + half * 8);
    v16bf b = load_b_frag(Brow + kk * 32 + half * 8);
    acc = wmma_bf16(a, b, acc);
  }
  int n = lane16;
  if (n < CONDD) {
    float bias = b_out[n];
    for (int i = 0; i < 8; ++i) {
      int m = mbase + i + 8 * half;
      out[(size_t)m * (NSTEP * CONDD) + (size_t)t * CONDD + n] =
          sigmoidf_(acc[i] + bias);
    }
  }
}

// ---- Host launch ------------------------------------------------------------
extern "C" void kernel_launch(void* const* d_in, const int* in_sizes, int n_in,
                              void* d_out, int out_size, void* d_ws, size_t ws_size,
                              hipStream_t stream) {
  const float* z      = (const float*)d_in[0];
  const float* cond   = (const float*)d_in[1];
  /* d_in[2] = iteration (unused, deterministic path) */
  const float* W_ih   = (const float*)d_in[3];
  const float* W_hh   = (const float*)d_in[4];
  const float* b_ih   = (const float*)d_in[5];
  const float* b_hh   = (const float*)d_in[6];
  const float* W_init = (const float*)d_in[7];
  const float* b_init = (const float*)d_in[8];
  const float* W_out  = (const float*)d_in[9];
  const float* b_out  = (const float*)d_in[10];

  auto al = [](size_t x) { return (x + 255) & ~(size_t)255; };
  char* w = (char*)d_ws;
  size_t o = 0;
  bf16_t* Whh_b   = (bf16_t*)(w + o); o += al((size_t)G3 * HID * 2);
  bf16_t* Wihz    = (bf16_t*)(w + o); o += al((size_t)G3 * Z1 * 2);
  bf16_t* Wihc    = (bf16_t*)(w + o); o += al((size_t)G3 * CPAD * 2);
  bf16_t* Winit_b = (bf16_t*)(w + o); o += al((size_t)HID * Z1 * 2);
  bf16_t* Wout_b  = (bf16_t*)(w + o); o += al((size_t)16 * HID * 2);
  bf16_t* zb      = (bf16_t*)(w + o); o += al((size_t)BSZ * Z1 * 2);
  float*  hf      = (float*) (w + o); o += al((size_t)BSZ * HID * 4);
  bf16_t* hb      = (bf16_t*)(w + o); o += al((size_t)BSZ * HID * 2);
  float*  giz     = (float*) (w + o); o += al((size_t)BSZ * G3 * 4);
  (void)ws_size; (void)in_sizes; (void)n_in; (void)out_size;

  k_cvt<<<(G3 * HID + 255) / 256, 256, 0, stream>>>(
      W_ih, W_hh, W_init, W_out, z, Whh_b, Wihz, Wihc, Winit_b, Wout_b, zb);

  k_init<<<(256 * 64) / 8, 256, 0, stream>>>(zb, Winit_b, b_init, hf, hb);
  k_giz<<<(256 * 192) / 8, 256, 0, stream>>>(zb, Wihz, b_ih, giz);

  for (int t = 0; t < NSTEP; ++t) {
    k_gru<<<BSZ / 16, 256, 0, stream>>>(cond, b_hh, giz, Whh_b, Wihc, hf, hb, t);
    k_out<<<256 / 8, 256, 0, stream>>>(hb, Wout_b, b_out, (float*)d_out, t);
  }
}
